// BiLSTM_CRF_73615739453778
// MI455X (gfx1250) — matrix operand order repaired and checked by reading, not compiled, not run on hardware
//
#include <hip/hip_runtime.h>
#include <hip/hip_bf16.h>

#define T_ 512
#define B_ 64
#define E_ 512
#define H1_ 512
#define H2_ 256
#define NTAG 48

typedef __hip_bfloat16 bf16;
typedef __attribute__((ext_vector_type(16))) __bf16 v16bf;
typedef __attribute__((ext_vector_type(8)))  float  v8f;

__device__ __forceinline__ float sigf(float x) { return 1.0f / (1.0f + expf(-x)); }

// ---------------------------------------------------------------------------
// WMMA fragment loaders for V_WMMA_F32_16X16X32_BF16 (wave32).
// A (16x32, row-major [M,K]): lane L owns row M=L&15; half=L>>4;
//   elements 0..7  -> K = k0 + half*8 + e        (contiguous 16B)
//   elements 8..15 -> K = k0 + 16 + half*8 + e-8 (contiguous 16B)
// B (32x16) fed from W[N,K] row-major (B[k,n] = W[n,k]): lane L owns col N=L&15;
//   same K mapping as A (elements 0..7 -> k0+half*8+e, 8..15 -> k0+16+half*8+e-8)
// ---------------------------------------------------------------------------
__device__ __forceinline__ v16bf load_frag_a(const bf16* rowbase, int ld, int k0) {
  const int lane = threadIdx.x & 31;
  const bf16* p = rowbase + (size_t)(lane & 15) * ld + k0 + ((lane >> 4) << 3);
  union { uint4 q[2]; v16bf v; } u;
  u.q[0] = *(const uint4*)p;
  u.q[1] = *(const uint4*)(p + 16);
  return u.v;
}

__device__ __forceinline__ v16bf load_frag_b(const bf16* rowbase, int ld, int k0) {
  const int lane = threadIdx.x & 31;
  const bf16* p = rowbase + (size_t)(lane & 15) * ld + k0 + ((lane >> 4) << 3);
  union { uint4 q[2]; v16bf v; } u;
  u.q[0] = *(const uint4*)p;
  u.q[1] = *(const uint4*)(p + 16);
  return u.v;
}

#define WMMA_BF16(a, b, c) __builtin_amdgcn_wmma_f32_16x16x32_bf16( \
    false, (a), false, (b), (short)0, (c), false, false)

// ---------------------------------------------------------------------------
// Generic WMMA GEMM: C[M,N] = A[M,K](bf16,row-major) * W[N,K]^T(bf16) + bias.
// 4 waves/block, each wave: one 16-row M tile x 64-wide N strip.
// Software-pipelined: fragments for k+32 are issued before the WMMAs of k,
// so loads overlap the matrix pipe (peeled prologue/epilogue, branch-free).
// ---------------------------------------------------------------------------
__global__ __launch_bounds__(128) void wmma_gemm_kernel(
    const bf16* __restrict__ A, const bf16* __restrict__ W,
    const float* __restrict__ bias, void* __restrict__ Cout,
    int M, int N, int K, int out_bf16) {
  const int lane = threadIdx.x & 31;
  const int wave = threadIdx.x >> 5;
  const int mt = blockIdx.x * 4 + wave;
  if (mt * 16 >= M) return;
  const int nt0 = blockIdx.y * 4;
  const int ntmax = N / 16;
  int n0 = nt0 + 0; bool v0 = n0 < ntmax; if (!v0) n0 = ntmax - 1;
  int n1 = nt0 + 1; bool v1 = n1 < ntmax; if (!v1) n1 = ntmax - 1;
  int n2 = nt0 + 2; bool v2 = n2 < ntmax; if (!v2) n2 = ntmax - 1;
  int n3 = nt0 + 3; bool v3 = n3 < ntmax; if (!v3) n3 = ntmax - 1;
  const bf16* W0 = W + (size_t)n0 * 16 * K;
  const bf16* W1 = W + (size_t)n1 * 16 * K;
  const bf16* W2 = W + (size_t)n2 * 16 * K;
  const bf16* W3 = W + (size_t)n3 * 16 * K;
  const bf16* Arow = A + (size_t)mt * 16 * K;
  v8f acc0 = {}, acc1 = {}, acc2 = {}, acc3 = {};
  v16bf a  = load_frag_a(Arow, K, 0);
  v16bf b0 = load_frag_b(W0, K, 0);
  v16bf b1 = load_frag_b(W1, K, 0);
  v16bf b2 = load_frag_b(W2, K, 0);
  v16bf b3 = load_frag_b(W3, K, 0);
#pragma unroll 2
  for (int k0 = 32; k0 < K; k0 += 32) {
    v16bf an  = load_frag_a(Arow, K, k0);
    v16bf c0v = load_frag_b(W0, K, k0);
    v16bf c1v = load_frag_b(W1, K, k0);
    v16bf c2v = load_frag_b(W2, K, k0);
    v16bf c3v = load_frag_b(W3, K, k0);
    acc0 = WMMA_BF16(a, b0, acc0);
    acc1 = WMMA_BF16(a, b1, acc1);
    acc2 = WMMA_BF16(a, b2, acc2);
    acc3 = WMMA_BF16(a, b3, acc3);
    a = an; b0 = c0v; b1 = c1v; b2 = c2v; b3 = c3v;
  }
  acc0 = WMMA_BF16(a, b0, acc0);
  acc1 = WMMA_BF16(a, b1, acc1);
  acc2 = WMMA_BF16(a, b2, acc2);
  acc3 = WMMA_BF16(a, b3, acc3);

  const int mb = mt * 16 + ((lane >> 4) << 3);
  const int nc = lane & 15;
  v8f accs[4] = {acc0, acc1, acc2, acc3};
  const bool vs[4] = {v0, v1, v2, v3};
#pragma unroll
  for (int q = 0; q < 4; ++q) {
    if (vs[q]) {
      const int n = (nt0 + q) * 16 + nc;
      const float bv = bias ? bias[n] : 0.0f;
#pragma unroll
      for (int r = 0; r < 8; ++r) {
        float v = accs[q][r] + bv;
        size_t idx = (size_t)(mb + r) * N + n;
        if (out_bf16) ((bf16*)Cout)[idx] = __float2bfloat16(v);
        else          ((float*)Cout)[idx] = v;
      }
    }
  }
}

// ---------------------------------------------------------------------------
// Multi-workgroup recurrent scan. Grid = 2*NW blocks of 512 threads:
// dir = blockIdx.x&1, wg = blockIdx.x>>1. Each WG owns hidden slice
// j in [wg*Hs, (wg+1)*Hs) and computes ALL FOUR gate quarters for it, so the
// cell update is WG-local; only h is exchanged through device-coherent L2.
// Per step: WMMA GEMM h[64,H] x Whh-slice -> gates in LDS (+G[t]); local
// barrier; cell update (c in registers) writes h slice to global; device-wide
// sense-free barrier (monotonic atomic counter, release/acquire fences,
// s_sleep spin) before the next step reads the full h.
// ---------------------------------------------------------------------------
template <int H, int NW>
__global__ __launch_bounds__(512) void lstm_scan_kernel(
    const bf16* __restrict__ Gf, const bf16* __restrict__ Gb,
    const bf16* __restrict__ Whhf, const bf16* __restrict__ Whhb,
    const float* __restrict__ h0, const float* __restrict__ c0,
    bf16* __restrict__ out, int out_ld,
    bf16* __restrict__ hstate, unsigned* __restrict__ counters) {
  constexpr int Hs = H / NW;                 // hidden slice per WG
  constexpr int NTL = 4 * Hs / 16;           // local gate n-tiles
  constexpr int TILES = 4 * NTL;             // x 4 M tiles
  constexpr int EPT = (B_ * Hs) / 512;       // cell elements per thread
  __shared__ alignas(16) float RgL[B_ * 4 * Hs];
  const int dir = blockIdx.x & 1;
  const int wg  = blockIdx.x >> 1;
  const bf16* G = dir ? Gb : Gf;
  const bf16* Whh = dir ? Whhb : Whhf;
  bf16* hs = hstate + (size_t)dir * B_ * H;
  unsigned* cnt = counters + dir;
  const int tid = threadIdx.x;
  const int lane = tid & 31;
  const int wave = tid >> 5;

  float creg[EPT];
#pragma unroll
  for (int i = 0; i < EPT; ++i) {
    const int el = tid + i * 512;
    const int b = el / Hs, jl = el - b * Hs;
    const int j = wg * Hs + jl;
    hs[b * H + j] = __float2bfloat16(h0[(size_t)dir * B_ * H + b * H + j]);
    creg[i] = c0[(size_t)dir * B_ * H + b * H + j];
  }

  unsigned tgt = NW;
  // device barrier 0: all initial h slices visible
  __builtin_amdgcn_fence(__ATOMIC_RELEASE, "agent");
  __syncthreads();
  if (tid == 0) {
    __hip_atomic_fetch_add(cnt, 1u, __ATOMIC_RELEASE, __HIP_MEMORY_SCOPE_AGENT);
    while (__hip_atomic_load(cnt, __ATOMIC_RELAXED, __HIP_MEMORY_SCOPE_AGENT) < tgt)
      __builtin_amdgcn_s_sleep(2);
  }
  __syncthreads();
  __builtin_amdgcn_fence(__ATOMIC_ACQUIRE, "agent");

  for (int s = 0; s < T_; ++s) {
    const int t = dir ? (T_ - 1 - s) : s;
    const bf16* Gt = G + (size_t)t * B_ * (4 * H);
    for (int tile = wave; tile < TILES; tile += 16) {
      const int mt = tile & 3;
      const int ntl = tile >> 2;
      const int q = ntl / (Hs / 16);
      const int rem = ntl - q * (Hs / 16);
      const int ngbase = q * H + wg * Hs + rem * 16;   // global gate column base
      v8f acc = {};
      const bf16* Arow = hs + mt * 16 * H;
      const bf16* Wrow = Whh + (size_t)ngbase * H;
#pragma unroll 2
      for (int k0 = 0; k0 < H; k0 += 32) {
        v16bf a = load_frag_a(Arow, H, k0);
        v16bf b = load_frag_b(Wrow, H, k0);
        acc = WMMA_BF16(a, b, acc);
      }
      const int nc = lane & 15;
      const int mb = mt * 16 + ((lane >> 4) << 3);
      const int nl = ntl * 16 + nc;
      const int ng = ngbase + nc;
#pragma unroll
      for (int r = 0; r < 8; ++r) {
        const int m = mb + r;
        RgL[m * (4 * Hs) + nl] =
            acc[r] + __bfloat162float(Gt[(size_t)m * (4 * H) + ng]);
      }
    }
    __syncthreads();

#pragma unroll
    for (int i = 0; i < EPT; ++i) {
      const int el = tid + i * 512;
      const int b = el / Hs, jl = el - b * Hs;
      const int j = wg * Hs + jl;
      const float gi = RgL[b * 4 * Hs + jl];
      const float gf = RgL[b * 4 * Hs + Hs + jl];
      const float gg = RgL[b * 4 * Hs + 2 * Hs + jl];
      const float go = RgL[b * 4 * Hs + 3 * Hs + jl];
      float c = sigf(gf) * creg[i] + sigf(gi) * tanhf(gg);
      float h = sigf(go) * tanhf(c);
      creg[i] = c;
      bf16 hb = __float2bfloat16(h);
      hs[b * H + j] = hb;
      out[((size_t)t * B_ + b) * out_ld + dir * H + j] = hb;
    }

    tgt += NW;
    __builtin_amdgcn_fence(__ATOMIC_RELEASE, "agent");
    __syncthreads();
    if (tid == 0) {
      __hip_atomic_fetch_add(cnt, 1u, __ATOMIC_RELEASE, __HIP_MEMORY_SCOPE_AGENT);
      while (__hip_atomic_load(cnt, __ATOMIC_RELAXED, __HIP_MEMORY_SCOPE_AGENT) < tgt)
        __builtin_amdgcn_s_sleep(2);
    }
    __syncthreads();
    __builtin_amdgcn_fence(__ATOMIC_ACQUIRE, "agent");
  }
}

// ---------------------------------------------------------------------------
__global__ void embed_gather_kernel(const int* __restrict__ seq,
                                    const float* __restrict__ table,
                                    bf16* __restrict__ x) {
  const int row = blockIdx.x;            // row = t*B + b
  const int t = row / B_;
  const int b = row - t * B_;
  const int v = seq[(size_t)b * T_ + t];
  const float* src = table + (size_t)v * E_;
  bf16* dst = x + (size_t)row * E_;
  for (int e = threadIdx.x; e < E_; e += blockDim.x)
    dst[e] = __float2bfloat16(src[e]);
}

__global__ void cvt_bf16_kernel(const float* __restrict__ src,
                                bf16* __restrict__ dst, int n) {
  int i = blockIdx.x * blockDim.x + threadIdx.x;
  if (i < n) dst[i] = __float2bfloat16(src[i]);
}

__global__ void badd_kernel(const float* __restrict__ a,
                            const float* __restrict__ b,
                            float* __restrict__ dst, int n) {
  int i = blockIdx.x * blockDim.x + threadIdx.x;
  if (i < n) dst[i] = a[i] + b[i];
}

__global__ void zero_u32_kernel(unsigned* __restrict__ p, int n) {
  int i = threadIdx.x;
  if (i < n) p[i] = 0u;
}

// ---------------------------------------------------------------------------
// CRF forward (log partition): single workgroup, alpha/trans in LDS.
// ---------------------------------------------------------------------------
__global__ __launch_bounds__(512) void crf_partition_kernel(
    const float* __restrict__ logits, const unsigned char* __restrict__ mask,
    const float* __restrict__ trans, const float* __restrict__ startv,
    const float* __restrict__ endv, float* __restrict__ logZ) {
  __shared__ float alpha[B_ * NTAG];
  __shared__ float beta[B_ * NTAG];
  __shared__ float trS[NTAG * NTAG];
  const int tid = threadIdx.x;
  for (int i = tid; i < NTAG * NTAG; i += blockDim.x) trS[i] = trans[i];
  for (int i = tid; i < B_ * NTAG; i += blockDim.x) {
    int b = i / NTAG, j = i - b * NTAG;
    alpha[i] = startv[j] + logits[(size_t)b * NTAG + j];   // t = 0
  }
  __syncthreads();
  for (int t = 1; t < T_; ++t) {
    const float* lt = logits + (size_t)t * B_ * NTAG;
    for (int i = tid; i < B_ * NTAG; i += blockDim.x) {
      int b = i / NTAG, j = i - b * NTAG;
      const float* arow = alpha + b * NTAG;
      float m = -3.4e38f;
      for (int q = 0; q < NTAG; ++q)
        m = fmaxf(m, arow[q] + trS[q * NTAG + j]);
      float ssum = 0.0f;
      for (int q = 0; q < NTAG; ++q)
        ssum += expf(arow[q] + trS[q * NTAG + j] - m);
      float nv = m + logf(ssum) + lt[b * NTAG + j];
      beta[i] = mask[(size_t)b * T_ + t] ? nv : arow[j];
    }
    __syncthreads();
    for (int i = tid; i < B_ * NTAG; i += blockDim.x) alpha[i] = beta[i];
    __syncthreads();
  }
  if (tid < B_) {
    const float* arow = alpha + tid * NTAG;
    float m = -3.4e38f;
    for (int q = 0; q < NTAG; ++q) m = fmaxf(m, arow[q] + endv[q]);
    float ssum = 0.0f;
    for (int q = 0; q < NTAG; ++q) ssum += expf(arow[q] + endv[q] - m);
    logZ[tid] = m + logf(ssum);
  }
}

// ---------------------------------------------------------------------------
__global__ void crf_joint_kernel(
    const float* __restrict__ logits, const int* __restrict__ tags,
    const unsigned char* __restrict__ mask, const float* __restrict__ trans,
    const float* __restrict__ startv, const float* __restrict__ endv,
    float* __restrict__ joint) {
  const int b = threadIdx.x;
  if (b >= B_) return;
  int tp = tags[(size_t)b * T_];
  float score = startv[tp];
  for (int t = 1; t < T_; ++t) {
    int tg = tags[(size_t)b * T_ + t];
    float mf = mask[(size_t)b * T_ + t] ? 1.0f : 0.0f;
    score += trans[tp * NTAG + tg] * mf;
    tp = tg;
  }
  for (int t = 0; t < T_ - 1; ++t) {
    int tg = tags[(size_t)b * T_ + t];
    float mf = mask[(size_t)b * T_ + t] ? 1.0f : 0.0f;
    score += logits[((size_t)t * B_ + b) * NTAG + tg] * mf;
  }
  int cnt = 0;
  for (int t = 0; t < T_; ++t) cnt += mask[(size_t)b * T_ + t] ? 1 : 0;
  int last = tags[(size_t)b * T_ + (cnt - 1)];
  float mfl = mask[(size_t)b * T_ + (T_ - 1)] ? 1.0f : 0.0f;
  score += endv[last] + logits[((size_t)(T_ - 1) * B_ + b) * NTAG + last] * mfl;
  joint[b] = score;
}

__global__ void crf_loss_kernel(const float* __restrict__ joint,
                                const float* __restrict__ logZ,
                                float* __restrict__ out) {
  __shared__ float s[B_];
  const int t = threadIdx.x;
  if (t < B_) s[t] = joint[t] - logZ[t];
  __syncthreads();
  if (t == 0) {
    float acc = 0.0f;
    for (int i = 0; i < B_; ++i) acc += s[i];
    out[0] = -acc;
  }
}

// ---------------------------------------------------------------------------
extern "C" void kernel_launch(void* const* d_in, const int* in_sizes, int n_in,
                              void* d_out, int out_size, void* d_ws, size_t ws_size,
                              hipStream_t stream) {
  (void)in_sizes; (void)n_in; (void)out_size; (void)ws_size;
  const int*   seq    = (const int*)d_in[0];
  const int*   tags   = (const int*)d_in[1];
  const unsigned char* mask = (const unsigned char*)d_in[2];
  const float* h0     = (const float*)d_in[3];
  const float* c0     = (const float*)d_in[4];
  const float* h1     = (const float*)d_in[5];
  const float* c1     = (const float*)d_in[6];
  const float* table  = (const float*)d_in[7];
  const float* lin_w  = (const float*)d_in[8];
  const float* lin_b  = (const float*)d_in[9];
  const float* trans  = (const float*)d_in[10];
  const float* startv = (const float*)d_in[11];
  const float* endv   = (const float*)d_in[12];
  const float* w_ih1f = (const float*)d_in[13];
  const float* w_hh1f = (const float*)d_in[14];
  const float* b_ih1f = (const float*)d_in[15];
  const float* b_hh1f = (const float*)d_in[16];
  const float* w_ih1b = (const float*)d_in[17];
  const float* w_hh1b = (const float*)d_in[18];
  const float* b_ih1b = (const float*)d_in[19];
  const float* b_hh1b = (const float*)d_in[20];
  const float* w_ih2f = (const float*)d_in[21];
  const float* w_hh2f = (const float*)d_in[22];
  const float* b_ih2f = (const float*)d_in[23];
  const float* b_hh2f = (const float*)d_in[24];
  const float* w_ih2b = (const float*)d_in[25];
  const float* w_hh2b = (const float*)d_in[26];
  const float* b_ih2b = (const float*)d_in[27];
  const float* b_hh2b = (const float*)d_in[28];

  char* ws = (char*)d_ws;
  size_t off = 0;
  auto carve = [&](size_t bytes) -> char* {
    char* p = ws + off;
    off += (bytes + 255) & ~(size_t)255;
    return p;
  };

  const size_t MB = (size_t)T_ * B_;          // 32768 rows
  bf16* xbf     = (bf16*)carve(MB * E_ * 2);
  bf16* w1f_ih  = (bf16*)carve((size_t)4 * H1_ * E_ * 2);
  bf16* w1f_hh  = (bf16*)carve((size_t)4 * H1_ * H1_ * 2);
  bf16* w1b_ih  = (bf16*)carve((size_t)4 * H1_ * E_ * 2);
  bf16* w1b_hh  = (bf16*)carve((size_t)4 * H1_ * H1_ * 2);
  bf16* w2f_ih  = (bf16*)carve((size_t)4 * H2_ * 2 * H1_ * 2);
  bf16* w2f_hh  = (bf16*)carve((size_t)4 * H2_ * H2_ * 2);
  bf16* w2b_ih  = (bf16*)carve((size_t)4 * H2_ * 2 * H1_ * 2);
  bf16* w2b_hh  = (bf16*)carve((size_t)4 * H2_ * H2_ * 2);
  bf16* linwbf  = (bf16*)carve((size_t)NTAG * (2 * H2_) * 2);
  float* bias1f = (float*)carve((size_t)4 * H1_ * 4);
  float* bias1b = (float*)carve((size_t)4 * H1_ * 4);
  float* bias2f = (float*)carve((size_t)4 * H2_ * 4);
  float* bias2b = (float*)carve((size_t)4 * H2_ * 4);
  bf16* out1    = (bf16*)carve(MB * (2 * H1_) * 2);
  bf16* out2    = (bf16*)carve(MB * (2 * H2_) * 2);
  float* logitsf= (float*)carve(MB * NTAG * 4);
  bf16* hstate  = (bf16*)carve((size_t)2 * B_ * H1_ * 2);
  unsigned* counters = (unsigned*)carve(256);
  float* logZ   = (float*)carve(B_ * 4);
  float* joint  = (float*)carve(B_ * 4);
  bf16* g1f     = (bf16*)carve(MB * 4 * H1_ * 2);   // reused as g2f region
  bf16* g1b     = (bf16*)carve(MB * 4 * H1_ * 2);   // reused as g2b region
  bf16* g2f     = g1f;
  bf16* g2b     = g1b;

  auto cvt = [&](const float* s, bf16* d, int n) {
    cvt_bf16_kernel<<<(n + 255) / 256, 256, 0, stream>>>(s, d, n);
  };
  cvt(w_ih1f, w1f_ih, 4 * H1_ * E_);
  cvt(w_hh1f, w1f_hh, 4 * H1_ * H1_);
  cvt(w_ih1b, w1b_ih, 4 * H1_ * E_);
  cvt(w_hh1b, w1b_hh, 4 * H1_ * H1_);
  cvt(w_ih2f, w2f_ih, 4 * H2_ * 2 * H1_);
  cvt(w_hh2f, w2f_hh, 4 * H2_ * H2_);
  cvt(w_ih2b, w2b_ih, 4 * H2_ * 2 * H1_);
  cvt(w_hh2b, w2b_hh, 4 * H2_ * H2_);
  cvt(lin_w, linwbf, NTAG * 2 * H2_);
  badd_kernel<<<(4 * H1_ + 255) / 256, 256, 0, stream>>>(b_ih1f, b_hh1f, bias1f, 4 * H1_);
  badd_kernel<<<(4 * H1_ + 255) / 256, 256, 0, stream>>>(b_ih1b, b_hh1b, bias1b, 4 * H1_);
  badd_kernel<<<(4 * H2_ + 255) / 256, 256, 0, stream>>>(b_ih2f, b_hh2f, bias2f, 4 * H2_);
  badd_kernel<<<(4 * H2_ + 255) / 256, 256, 0, stream>>>(b_ih2b, b_hh2b, bias2b, 4 * H2_);

  embed_gather_kernel<<<T_ * B_, 128, 0, stream>>>(seq, table, xbf);

  // Layer-1 input projections: [32768,512] x [512,2048]
  wmma_gemm_kernel<<<dim3((int)MB / 64, (4 * H1_) / 64), 128, 0, stream>>>(
      xbf, w1f_ih, bias1f, g1f, (int)MB, 4 * H1_, E_, 1);
  wmma_gemm_kernel<<<dim3((int)MB / 64, (4 * H1_) / 64), 128, 0, stream>>>(
      xbf, w1b_ih, bias1b, g1b, (int)MB, 4 * H1_, E_, 1);

  // Layer-1 recurrent scan: 16 WGs per direction, device-barrier per step
  zero_u32_kernel<<<1, 32, 0, stream>>>(counters, 2);
  lstm_scan_kernel<H1_, 16><<<32, 512, 0, stream>>>(
      g1f, g1b, w1f_hh, w1b_hh, h0, c0, out1, 2 * H1_, hstate, counters);

  // Layer-2 input projections: [32768,1024] x [1024,1024]  (reuses G region)
  wmma_gemm_kernel<<<dim3((int)MB / 64, (4 * H2_) / 64), 128, 0, stream>>>(
      out1, w2f_ih, bias2f, g2f, (int)MB, 4 * H2_, 2 * H1_, 1);
  wmma_gemm_kernel<<<dim3((int)MB / 64, (4 * H2_) / 64), 128, 0, stream>>>(
      out1, w2b_ih, bias2b, g2b, (int)MB, 4 * H2_, 2 * H1_, 1);

  // Layer-2 recurrent scan
  zero_u32_kernel<<<1, 32, 0, stream>>>(counters, 2);
  lstm_scan_kernel<H2_, 16><<<32, 512, 0, stream>>>(
      g2f, g2b, w2f_hh, w2b_hh, h1, c1, out2, 2 * H2_, hstate, counters);

  // Logits: [32768,512] x [512,48] -> f32
  wmma_gemm_kernel<<<dim3((int)MB / 64, 1), 128, 0, stream>>>(
      out2, linwbf, lin_b, logitsf, (int)MB, NTAG, 2 * H2_, 0);

  // CRF
  crf_partition_kernel<<<1, 512, 0, stream>>>(logitsf, mask, trans, startv, endv, logZ);
  crf_joint_kernel<<<1, 64, 0, stream>>>(logitsf, tags, mask, trans, startv, endv, joint);
  crf_loss_kernel<<<1, 64, 0, stream>>>(joint, logZ, (float*)d_out);
}
